// PHM8Linear_37941741093226
// MI455X (gfx1250) — matrix operand order repaired and checked by reading, not compiled, not run on hardware
//
#include <hip/hip_runtime.h>
#include <hip/hip_bf16.h>

// PHM8 linear: out[t, r] = sum_k x_flat[t, k] * H[r, k] + bias[r]
//   H[r, k] = sum_i A[i, r/64, k/64] * S[i, r%64, k%64]   (512 x 512)
//   M = 16*8192 = 131072, N = 512, K = 512
//
// MI455X (gfx1250, wave32) strategy:
//   kernel 1: build H, split fp32 -> f16 hi/lo pair in d_ws (2 * 512KB)
//   kernel 2: WMMA GEMM, f32 accum, 3-term f16 split (xh*Hh + xh*Hl + xl*Hh)
//             block tile 256x128, 8 waves stacked along M,
//             wave tile 32x128 (2x8 accumulators), BK = 32, 48 WMMA/k-step.
//   All global accesses: loop-invariant per-lane VGPR offset + immediate
//   IOFFSET, scalar k-advance. A stream (HBM) and B stream (L2) both
//   software-pipelined one step ahead; x prefetched at WGP scope.

typedef __attribute__((ext_vector_type(16))) _Float16 v16h;
typedef __attribute__((ext_vector_type(8)))  float    v8f;

#define KDIM 512
#define RDIM 512

// ---------------- kernel 1: build H hi/lo (512x512) ----------------
__global__ void phm8_build_h(const float* __restrict__ A,   // (8,8,8)  [i,a,c]
                             const float* __restrict__ S,   // (8,64,64)[i,b,d]
                             _Float16* __restrict__ Hhi,
                             _Float16* __restrict__ Hlo) {
    int idx = blockIdx.x * blockDim.x + threadIdx.x;  // 0 .. 512*512-1
    int r = idx >> 9;          // row: r = a*64 + b
    int k = idx & 511;         // col: k = c*64 + d
    int a = r >> 6, b = r & 63;
    int c = k >> 6, d = k & 63;
    float h = 0.0f;
#pragma unroll
    for (int i = 0; i < 8; ++i)
        h += A[(i * 8 + a) * 8 + c] * S[i * 4096 + b * 64 + d];
    _Float16 hi = (_Float16)h;
    _Float16 lo = (_Float16)(h - (float)hi);
    Hhi[idx] = hi;
    Hlo[idx] = lo;
}

// ---------------- kernel 2: WMMA GEMM ----------------
__global__ __launch_bounds__(256)
void phm8_gemm(const float*    __restrict__ X,     // (131072, 512) fp32
               const _Float16* __restrict__ Hhi,   // (512, 512) f16
               const _Float16* __restrict__ Hlo,   // (512, 512) f16
               const float*    __restrict__ bias,  // (512,)
               float*          __restrict__ out) { // (131072, 512) fp32
    const int lane  = threadIdx.x & 31;
    const int wave  = threadIdx.x >> 5;
    const int m_base = blockIdx.y * 256 + wave * 32;   // 8 waves stacked along M
    const int r_base = blockIdx.x * 128;               // every wave spans all 128 cols

    const int lrow  = lane & 15;                // row (A) / column (B) within fragment
    const int lhalf = lane >> 4;                // 0: lanes 0-15, 1: lanes 16-31
    const int akoff = lhalf * 16;               // A-frag: K chunk  {akoff .. akoff+15}
    const int bkoff = lhalf * 16;               // B-frag: K chunk  {bkoff .. bkoff+15}

    // Loop-invariant per-lane byte offsets (single VGPR each); mi / chunk / ni
    // displacements become compile-time IOFFSET immediates.
    const int aoff = ((m_base + lrow) * KDIM + akoff) * 4;   // < 2^31
    const int boff = ((r_base + lrow) * KDIM + bkoff) * 2;

    const char* Xb  = (const char*)X;
    const char* Hhb = (const char*)Hhi;
    const char* Hlb = (const char*)Hlo;

    v8f acc[2][8] = {};                         // 16 accumulators of 8 VGPRs (f32)

    // ---- prologue: A vectors for k0=0, B fragment pair for (k0=0, ni=0) ----
    v8f nx00 = *(const v8f*)(Xb + aoff);
    v8f nx01 = *(const v8f*)(Xb + aoff + 32);
    v8f nx10 = *(const v8f*)(Xb + aoff + 32768);
    v8f nx11 = *(const v8f*)(Xb + aoff + 32768 + 32);
    v16h bh_c = *(const v16h*)(Hhb + boff);
    v16h bl_c = *(const v16h*)(Hlb + boff);

#pragma unroll 1
    for (int k0 = 0; k0 < KDIM; k0 += 32) {
        // Scalar (SALU) k-advance; next-step bases clamp to 0 on last iter so
        // the pipelined loads stay in-bounds (their data is simply unused).
        const int   kn    = (k0 + 32 < KDIM) ? (k0 + 32) : 0;
        const char* xk_n  = Xb  + (size_t)kn * 4;
        const char* hk_h  = Hhb + (size_t)k0 * 2;
        const char* hk_l  = Hlb + (size_t)k0 * 2;
        const char* hk_hn = Hhb + (size_t)kn * 2;
        const char* hk_ln = Hlb + (size_t)kn * 2;

        // ---- split current A vectors into f16 hi/lo in contiguous k order (k = akoff + s) ----
        // hi packed first; lo = fma(-1, hi, x) references the packed halves so
        // v_fma_mix can consume them via op_sel (no rematerialized scalar cvt).
        v16h ahi[2], alo[2];
        {
            v8f c0[2] = { nx00, nx10 };
            v8f c1[2] = { nx01, nx11 };
#pragma unroll
            for (int mi = 0; mi < 2; ++mi) {
                v16h h;
#pragma unroll
                for (int j = 0; j < 8; ++j) {
                    h[j]     = (_Float16)c0[mi][j];
                    h[8 + j] = (_Float16)c1[mi][j];
                }
                v16h l;
#pragma unroll
                for (int j = 0; j < 8; ++j) {
                    l[j]     = (_Float16)__builtin_fmaf(-1.0f, (float)h[j],     c0[mi][j]);
                    l[8 + j] = (_Float16)__builtin_fmaf(-1.0f, (float)h[8 + j], c1[mi][j]);
                }
                ahi[mi] = h;
                alo[mi] = l;
            }
        }

        // ---- issue next k-step's A loads early (fly under the WMMAs) ----
        __builtin_prefetch(xk_n + aoff + 128, 0, 3);   // WGP-scope global_prefetch_b8
        nx00 = *(const v8f*)(xk_n + aoff);
        nx01 = *(const v8f*)(xk_n + aoff + 32);
        nx10 = *(const v8f*)(xk_n + aoff + 32768);
        nx11 = *(const v8f*)(xk_n + aoff + 32768 + 32);

        // ---- B double-buffer: load ni+1 (or ni=0 of next k) under ni's WMMAs ----
#pragma unroll
        for (int ni = 0; ni < 8; ++ni) {
            v16h bh_x, bl_x;
            if (ni < 7) {
                bh_x = *(const v16h*)(hk_h + boff + (ni + 1) * 16384);
                bl_x = *(const v16h*)(hk_l + boff + (ni + 1) * 16384);
            } else {
                bh_x = *(const v16h*)(hk_hn + boff);
                bl_x = *(const v16h*)(hk_ln + boff);
            }
#pragma unroll
            for (int mi = 0; mi < 2; ++mi) {
                acc[mi][ni] = __builtin_amdgcn_wmma_f32_16x16x32_f16(
                    false, ahi[mi], false, bh_c, (short)0, acc[mi][ni], false, false);
                acc[mi][ni] = __builtin_amdgcn_wmma_f32_16x16x32_f16(
                    false, ahi[mi], false, bl_c, (short)0, acc[mi][ni], false, false);
                acc[mi][ni] = __builtin_amdgcn_wmma_f32_16x16x32_f16(
                    false, alo[mi], false, bh_c, (short)0, acc[mi][ni], false, false);
            }
            bh_c = bh_x;
            bl_c = bl_x;
        }
    }

    // ---- epilogue: C layout -> row m = base + lhalf*8 + j, col = lane&15 ----
    // All stores: per-lane VGPR offset + immediate IOFFSET.
    char* ob = (char*)out;
    const int obase = ((m_base + lhalf * 8) * RDIM + r_base + lrow) * 4;
    const char* bb = (const char*)bias + (size_t)r_base * 4;

    float bv[8];
#pragma unroll
    for (int ni = 0; ni < 8; ++ni)
        bv[ni] = *(const float*)(bb + lrow * 4 + ni * 64);

#pragma unroll
    for (int ni = 0; ni < 8; ++ni) {
#pragma unroll
        for (int mi = 0; mi < 2; ++mi) {
#pragma unroll
            for (int j = 0; j < 8; ++j) {
                *(float*)(ob + obase + ni * 64 + (mi * 16 + j) * (RDIM * 4)) =
                    acc[mi][ni][j] + bv[ni];
            }
        }
    }
}

extern "C" void kernel_launch(void* const* d_in, const int* in_sizes, int n_in,
                              void* d_out, int out_size, void* d_ws, size_t ws_size,
                              hipStream_t stream) {
    (void)in_sizes; (void)n_in; (void)out_size; (void)ws_size;
    const float* X    = (const float*)d_in[0];   // (16, 8192, 64, 8)
    const float* A    = (const float*)d_in[1];   // (8, 8, 8)
    const float* S    = (const float*)d_in[2];   // (8, 64, 64)
    const float* bias = (const float*)d_in[3];   // (512,)
    float* out = (float*)d_out;                  // (16, 8192, 64, 8) flat

    _Float16* Hhi = (_Float16*)d_ws;             // 512*512 f16 = 512 KB
    _Float16* Hlo = Hhi + 512 * 512;             // another 512 KB

    // Build H (hi/lo f16 split): 512*512 elements, 256 threads/block
    phm8_build_h<<<dim3(1024), dim3(256), 0, stream>>>(A, S, Hhi, Hlo);

    // GEMM: grid = (N tiles = 4 fastest for x L2 reuse, M tiles = 512)
    phm8_gemm<<<dim3(4, 512), dim3(256), 0, stream>>>(X, Hhi, Hlo, bias, out);
}
